// MultiHeadAttentionLayer_26929444946498
// MI455X (gfx1250) — matrix operand order repaired and checked by reading, not compiled, not run on hardware
//
#include <hip/hip_runtime.h>

// ---------------------------------------------------------------------------
// Multi-head attention forward for MI455X (gfx1250, wave32, WMMA).
// Outputs: x (B,S,HID) fp32  then  attention (B,H,S,S) fp32, concatenated.
// Workspace: Q,K,V,ctx stored as f16 (4 * 16 MB = 64 MB).
// Pipeline:
//   1) 3x gemm_bias_wmma : Q/K/V projections (fp32 in -> f16 ws)
//   2) attn_scores_softmax_wmma : QK^T/8 + mask + row softmax fused,
//      scores kept in VGPRs (block = 16 rows x 1024 cols) -> single write.
//      K tiles staged via TDM tensor_load_to_lds when available.
//   3) attn_v_wmma : P @ V -> f16 ctx (coalesced V transpose through LDS)
//   4) gemm_bias_wmma : output projection (f16 ws -> fp32 d_out)
// ---------------------------------------------------------------------------

#define B_     8
#define S_     1024
#define HID_   1024
#define HEADS_ 16
#define DHEAD_ 64

#if defined(__has_builtin)
#  if __has_builtin(__builtin_amdgcn_tensor_load_to_lds)
#    define HAVE_TDM 1
#  endif
#  if __has_builtin(__builtin_amdgcn_global_load_async_to_lds_b128)
#    define HAVE_ASYNC 1
#  endif
#endif
#ifndef HAVE_TDM
#  define HAVE_TDM 0
#endif
#ifndef HAVE_ASYNC
#  define HAVE_ASYNC 0
#endif

typedef __attribute__((ext_vector_type(16))) _Float16 v16h;
typedef __attribute__((ext_vector_type(8)))  float    v8f;
typedef __attribute__((ext_vector_type(4)))  unsigned int v4u;
typedef __attribute__((ext_vector_type(8)))  int      v8i;
typedef __attribute__((ext_vector_type(4)))  int      v4i;

union FragAB { v16h v; unsigned int u[8]; };

__device__ __forceinline__ unsigned short f2h_u16(float f) {
  _Float16 h = (_Float16)f;               // v_cvt_f16_f32 (RNE)
  return __builtin_bit_cast(unsigned short, h);
}
__device__ __forceinline__ unsigned int pack_h2(float lo, float hi) {
  return (unsigned int)f2h_u16(lo) | ((unsigned int)f2h_u16(hi) << 16);
}

// A-fragment (16x32 f16): lane = half*16 + m;  u32 i holds k-pair
//   kpair(i) = i + 4*((i>>2) + half)     (ISA 7.12.2, 16-bit A layout)
__device__ __forceinline__ void load_frag_a(const unsigned int* lds, int mbase,
                                            int stride, int l16, int half,
                                            FragAB& a) {
  const unsigned int* row = lds + (mbase + l16) * stride;
#pragma unroll
  for (int i = 0; i < 8; ++i) a.u[i] = row[i + 4 * ((i >> 2) + half)];
}
// B-fragment (32x16 f16): lane = half*16 + n;  u32 i holds k-pair
//   kpair(i) = i + 4*((i>>2) + half)     (same k order as A fragment)
__device__ __forceinline__ void load_frag_b(const unsigned int* lds, int nbase,
                                            int stride, int l16, int half,
                                            FragAB& b) {
  const unsigned int* row = lds + (nbase + l16) * stride;
#pragma unroll
  for (int i = 0; i < 8; ++i) b.u[i] = row[i + 4 * ((i >> 2) + half)];
}

__device__ __forceinline__ v8f wmma_f16(const FragAB& a, const FragAB& b, v8f c) {
  return __builtin_amdgcn_wmma_f32_16x16x32_f16(
      /*neg_a=*/false, a.v, /*neg_b=*/false, b.v,
      /*c_mod=*/(short)0, c, /*reuse_a=*/false, /*reuse_b=*/false);
}

// ---- gfx1250 async / tensor data-mover helpers (guarded) ------------------
#if HAVE_ASYNC
__device__ __forceinline__ void async_b128(const void* g, void* l) {
  // builtin expects v4i* params (16-byte granules): (global src, lds dst)
  __builtin_amdgcn_global_load_async_to_lds_b128((v4i*)g, (v4i*)l, 0, 0);
}
__device__ __forceinline__ void async_wait() {
#  if __has_builtin(__builtin_amdgcn_s_wait_asynccnt)
  __builtin_amdgcn_s_wait_asynccnt(0);
#  else
  asm volatile("s_wait_asynccnt 0x0" ::: "memory");
#  endif
}
#endif

#if HAVE_TDM
// 2D f16 tile load: tile_rows x tile_cols (elements), global row stride
// row_stride_elems, packed contiguously into LDS at lds_off.  D# built per
// ISA 8.3/8.4 (count=1, type=2 image, data_size=1 -> 2 bytes).  Groups 2/3
// zero (tile_dim2=0 -> unused; 2D tile).
__device__ __forceinline__ void tdm_load_2d_f16(const void* gaddr,
                                                unsigned lds_off, int tile_rows,
                                                int tile_cols,
                                                int row_stride_elems) {
  const unsigned long long ga = (unsigned long long)(size_t)gaddr;
  v4u g0;
  g0[0] = 1u;                                       // count=1 (valid user D#)
  g0[1] = lds_off;                                  // lds_addr
  g0[2] = (unsigned)(ga & 0xffffffffu);             // global_addr[31:0]
  g0[3] = (unsigned)((ga >> 32) & 0x1ffffffu) | (2u << 30);  // [56:32] | type=2
  v8i g1;
  g1[0] = (int)(1u << 16);                          // data_size=1 (2 bytes)
  g1[1] = (int)(((unsigned)tile_cols & 0xffffu) << 16);  // tensor_dim0[15:0]
  g1[2] = (int)(((unsigned)tile_rows & 0xffffu) << 16);  // tensor_dim1[15:0]
  g1[3] = (int)(((unsigned)tile_cols & 0xffffu) << 16);  // tile_dim0
  g1[4] = (int)((unsigned)tile_rows & 0xffffu);          // tile_dim1
  g1[5] = row_stride_elems;                         // tensor_dim0_stride[31:0]
  g1[6] = 0;
  g1[7] = 0;
  v4i z4 = {0, 0, 0, 0};
  v8i z8 = {0, 0, 0, 0, 0, 0, 0, 0};
  __builtin_amdgcn_tensor_load_to_lds(g0, g1, z4, z4, z8, 0);
}
#endif

// ---------------------------------------------------------------------------
// Generic C[M,N] = A[M,K] @ W[N,K]^T + bias.  Block tile 64x64, BK=32,
// 8 waves as 4(m) x 2(n), each wave: one 16x32 strip (two f32 accumulators).
// ---------------------------------------------------------------------------
template <bool A_IS_F16, bool OUT_IS_F16>
__global__ __launch_bounds__(256)
void gemm_bias_wmma(const void* __restrict__ Aq, const float* __restrict__ W,
                    const float* __restrict__ bias, void* __restrict__ Cq,
                    int M, int N, int K) {
  __shared__ unsigned int ldsA[64 * 16];   // [m][kpair]  (64x32 f16)
  __shared__ unsigned int ldsB[64 * 16];   // [n][kpair]  (64x32 f16)

  const int tid  = threadIdx.x;
  const int lane = tid & 31, wave = tid >> 5;
  const int half = lane >> 4, l16 = lane & 15;
  const int wm = (wave & 3) * 16;
  const int wn = (wave >> 2) * 32;
  const int m0 = blockIdx.y * 64;
  const int n0 = blockIdx.x * 64;
  (void)M;

  v8f c0 = {}; v8f c1 = {};

  for (int k0 = 0; k0 < K; k0 += 32) {
    // ---- stage A tile ----
    if constexpr (A_IS_F16) {
      const unsigned int* Au = (const unsigned int*)Aq;
#if HAVE_ASYNC
      for (int q = tid; q < 64 * 4; q += 256) {        // 256 b128 segments
        const int r = q >> 2, seg = q & 3;
        async_b128(Au + (size_t)(m0 + r) * (K / 2) + (k0 >> 1) + seg * 4,
                   &ldsA[r * 16 + seg * 4]);
      }
#else
      for (int p = tid; p < 1024; p += 256) {
        const int r = p >> 4, kp = p & 15;
        ldsA[p] = Au[(size_t)(m0 + r) * (K / 2) + (k0 >> 1) + kp];
      }
#endif
    } else {
      const float* Af = (const float*)Aq;
      for (int p = tid; p < 1024; p += 256) {
        const int r = p >> 4, kp = p & 15;
        const float2 f2 =
            *(const float2*)(Af + (size_t)(m0 + r) * K + k0 + 2 * kp);
        ldsA[p] = pack_h2(f2.x, f2.y);
      }
    }
    // ---- stage W tile (fp32 -> f16 convert) ----
    for (int p = tid; p < 1024; p += 256) {
      const int r = p >> 4, kp = p & 15;
      const float2 w2 = *(const float2*)(W + (size_t)(n0 + r) * K + k0 + 2 * kp);
      ldsB[p] = pack_h2(w2.x, w2.y);
    }
#if HAVE_ASYNC
    if constexpr (A_IS_F16) async_wait();
#endif
    __syncthreads();

    FragAB a, b0, b1;
    load_frag_a(ldsA, wm, 16, l16, half, a);
    load_frag_b(ldsB, wn, 16, l16, half, b0);
    load_frag_b(ldsB, wn + 16, 16, l16, half, b1);
    c0 = wmma_f16(a, b0, c0);
    c1 = wmma_f16(a, b1, c1);
    __syncthreads();
  }

  const int colA = n0 + wn + l16;
  const int colB = colA + 16;
  const float ba = bias[colA], bb = bias[colB];
#pragma unroll
  for (int j = 0; j < 8; ++j) {
    const int row = m0 + wm + j + 8 * half;   // C/D layout: M = j + 8*half
    const float va = c0[j] + ba;
    const float vb = c1[j] + bb;
    if constexpr (OUT_IS_F16) {
      unsigned short* Cu = (unsigned short*)Cq;
      Cu[(size_t)row * N + colA] = f2h_u16(va);
      Cu[(size_t)row * N + colB] = f2h_u16(vb);
    } else {
      float* Cf = (float*)Cq;
      Cf[(size_t)row * N + colA] = va;
      Cf[(size_t)row * N + colB] = vb;
    }
  }
}

// ---------------------------------------------------------------------------
// Fused scores + softmax.  Block = 16 query rows x full 1024 key cols,
// per (b,h).  Scores live in VGPRs: wave w owns cols [w*32, w*32+32) of each
// 256-col chunk -> 8 accumulators (16x16 each) per wave.  Row softmax via
// shfl_xor within 16-lane half-groups + small LDS cross-wave reduce.
// K tile staged by TDM (one tensor_load_to_lds per 256x64 chunk) if present.
// ---------------------------------------------------------------------------
__global__ __launch_bounds__(256)
void attn_scores_softmax_wmma(const unsigned int* __restrict__ Qu,
                              const unsigned int* __restrict__ Ku,
                              const int* __restrict__ mask,
                              float* __restrict__ attn) {
  __shared__ unsigned int ldsQ[16 * 32];    // [m][kpair]   16x64 f16  (2 KB)
  __shared__ unsigned int ldsK[256 * 32];   // [n][kpair]  256x64 f16 (32 KB)
  __shared__ float redbuf[16 * 8];          // [row][wave] partials
  __shared__ float rowstat[16];             // final per-row max / sum

  const int tid  = threadIdx.x;
  const int lane = tid & 31, wave = tid >> 5;
  const int half = lane >> 4, l16 = lane & 15;
  const int bh = blockIdx.y, b = bh >> 4, h = bh & 15;
  const int m0 = blockIdx.x * 16;           // query-row tile
  const int hk = h * (DHEAD_ / 2);          // u32 offset of head within row

  // ---- stage Q tile once (16 rows x 32 u32) ----
#if HAVE_ASYNC
  for (int q = tid; q < 16 * 8; q += 256) {
    const int r = q >> 3, seg = q & 7;
    async_b128(Qu + (size_t)(b * S_ + m0 + r) * (HID_ / 2) + hk + seg * 4,
               &ldsQ[r * 32 + seg * 4]);
  }
#else
  for (int p = tid; p < 16 * 32; p += 256) {
    const int r = p >> 5, kp = p & 31;
    ldsQ[p] = Qu[(size_t)(b * S_ + m0 + r) * (HID_ / 2) + hk + kp];
  }
#endif

  v8f acc[8];
  {
    v8f z = {};
#pragma unroll
    for (int i = 0; i < 8; ++i) acc[i] = z;
  }

  for (int c = 0; c < 4; ++c) {
    const int n0 = c * 256;
    __syncthreads();                        // all waves done with prev ldsK
#if HAVE_TDM
    if (wave == 0) {
      tdm_load_2d_f16(Ku + (size_t)(b * S_ + n0) * (HID_ / 2) + hk,
                      (unsigned)(unsigned long long)(size_t)(void*)ldsK,
                      /*rows=*/256, /*cols=*/DHEAD_, /*stride=*/HID_);
      __builtin_amdgcn_s_wait_tensorcnt(0);
    }
#if HAVE_ASYNC
    if (c == 0) async_wait();               // Q-tile async completion
#endif
#elif HAVE_ASYNC
    for (int q = tid; q < 256 * 8; q += 256) {
      const int r = q >> 3, seg = q & 7;
      async_b128(Ku + (size_t)(b * S_ + n0 + r) * (HID_ / 2) + hk + seg * 4,
                 &ldsK[r * 32 + seg * 4]);
    }
    async_wait();
#else
    for (int p = tid; p < 256 * 32; p += 256) {
      const int r = p >> 5, kp = p & 31;
      ldsK[p] = Ku[(size_t)(b * S_ + n0 + r) * (HID_ / 2) + hk + kp];
    }
#endif
    __syncthreads();
#pragma unroll
    for (int ks = 0; ks < 2; ++ks) {        // K=64 -> two 32-deep WMMA steps
      FragAB a;
      load_frag_a(ldsQ + ks * 16, 0, 32, l16, half, a);
#pragma unroll
      for (int t = 0; t < 2; ++t) {
        FragAB bf;
        load_frag_b(ldsK + ks * 16, wave * 32 + t * 16, 32, l16, half, bf);
        acc[c * 2 + t] = wmma_f16(a, bf, acc[c * 2 + t]);
      }
    }
  }

  // ---- scale + mask; per-row (j) max over this lane's 8 columns ----
  const float scale = 0.125f;               // 1/sqrt(64)
  float sc[8][8];                           // [acc][j]
  float rmax[8];
#pragma unroll
  for (int j = 0; j < 8; ++j) rmax[j] = -3.4e38f;
#pragma unroll
  for (int i = 0; i < 8; ++i) {
    const int c = i >> 1, t = i & 1;
    const int col = c * 256 + wave * 32 + t * 16 + l16;
    const int mv = mask[b * S_ + col];
#pragma unroll
    for (int j = 0; j < 8; ++j) {
      const float v = mv ? acc[i][j] * scale : -1e10f;
      sc[i][j] = v;
      rmax[j] = fmaxf(rmax[j], v);
    }
  }
#pragma unroll
  for (int j = 0; j < 8; ++j) {
    float m = rmax[j];
#pragma unroll
    for (int off = 1; off <= 8; off <<= 1) m = fmaxf(m, __shfl_xor(m, off, 32));
    rmax[j] = m;
  }
  if (l16 == 0) {
#pragma unroll
    for (int j = 0; j < 8; ++j) redbuf[(j + 8 * half) * 8 + wave] = rmax[j];
  }
  __syncthreads();
  if (tid < 16) {
    float m = redbuf[tid * 8];
#pragma unroll
    for (int w = 1; w < 8; ++w) m = fmaxf(m, redbuf[tid * 8 + w]);
    rowstat[tid] = m;
  }
  __syncthreads();
  float fmx[8];
#pragma unroll
  for (int j = 0; j < 8; ++j) fmx[j] = rowstat[j + 8 * half];
  __syncthreads();                          // rowstat consumed before reuse

  // ---- exp + per-row sum (same reduction path) ----
  float rsum[8];
#pragma unroll
  for (int j = 0; j < 8; ++j) rsum[j] = 0.f;
#pragma unroll
  for (int i = 0; i < 8; ++i) {
#pragma unroll
    for (int j = 0; j < 8; ++j) {
      sc[i][j] = __expf(sc[i][j] - fmx[j]);
      rsum[j] += sc[i][j];
    }
  }
#pragma unroll
  for (int j = 0; j < 8; ++j) {
    float s = rsum[j];
#pragma unroll
    for (int off = 1; off <= 8; off <<= 1) s += __shfl_xor(s, off, 32);
    rsum[j] = s;
  }
  if (l16 == 0) {
#pragma unroll
    for (int j = 0; j < 8; ++j) redbuf[(j + 8 * half) * 8 + wave] = rsum[j];
  }
  __syncthreads();
  if (tid < 16) {
    float s = 0.f;
#pragma unroll
    for (int w = 0; w < 8; ++w) s += redbuf[tid * 8 + w];
    rowstat[tid] = s;
  }
  __syncthreads();
  float inv[8];
#pragma unroll
  for (int j = 0; j < 8; ++j) inv[j] = 1.0f / rowstat[j + 8 * half];

  // ---- single normalized write of the attention tile ----
#pragma unroll
  for (int i = 0; i < 8; ++i) {
    const int c = i >> 1, t = i & 1;
    const int col = c * 256 + wave * 32 + t * 16 + l16;
#pragma unroll
    for (int j = 0; j < 8; ++j) {
      const int row = m0 + j + 8 * half;
      attn[((size_t)bh * S_ + row) * S_ + col] = sc[i][j] * inv[j];
    }
  }
}

// ---------------------------------------------------------------------------
// ctx[b,sq,h*64+d] = sum_sk P[b,h,sq,sk] * V[b,sk,h*64+d].  K=1024, N=64.
// V staged with coalesced u32 row loads + b16 LDS transpose (padded rows).
// ---------------------------------------------------------------------------
__global__ __launch_bounds__(256)
void attn_v_wmma(const float* __restrict__ attn,
                 const unsigned int* __restrict__ Vu,
                 unsigned short* __restrict__ ctx) {
  __shared__ unsigned int ldsA[64 * 16];   // P tile [sq][kpair], stride 16
  __shared__ unsigned int ldsB[64 * 18];   // V tile [d][kpair],  stride 18 (pad)

  const int tid  = threadIdx.x;
  const int lane = tid & 31, wave = tid >> 5;
  const int half = lane >> 4, l16 = lane & 15;
  const int wm = (wave & 3) * 16, wn = (wave >> 2) * 32;
  const int bh = blockIdx.y, b = bh >> 4, h = bh & 15;
  const int m0 = blockIdx.x * 64;

  v8f c0 = {}; v8f c1 = {};

  for (int k0 = 0; k0 < S_; k0 += 32) {
    for (int p = tid; p < 1024; p += 256) {
      // A: attention rows (fp32 -> f16)
      const int r = p >> 4, kp = p & 15;
      const float2 a2 = *(const float2*)(
          attn + ((size_t)bh * S_ + m0 + r) * S_ + k0 + 2 * kp);
      ldsA[p] = pack_h2(a2.x, a2.y);
      // B: coalesced V row load (u32 = 2 d-values), transpose into [d][k]
      const int kk = p >> 5;        // seq row within chunk  (0..31)
      const int dp = p & 31;        // d pair                (0..31)
      const unsigned int w =
          Vu[(size_t)(b * S_ + k0 + kk) * (HID_ / 2) + h * (DHEAD_ / 2) + dp];
      unsigned short* s = (unsigned short*)ldsB;
      s[(2 * dp) * 36 + kk]     = (unsigned short)(w & 0xffffu);
      s[(2 * dp + 1) * 36 + kk] = (unsigned short)(w >> 16);
    }
    __syncthreads();
    FragAB a, b0, b1;
    load_frag_a(ldsA, wm, 16, l16, half, a);
    load_frag_b(ldsB, wn, 18, l16, half, b0);
    load_frag_b(ldsB, wn + 16, 18, l16, half, b1);
    c0 = wmma_f16(a, b0, c0);
    c1 = wmma_f16(a, b1, c1);
    __syncthreads();
  }

  const int colA = wn + l16;
  const int colB = colA + 16;
#pragma unroll
  for (int j = 0; j < 8; ++j) {
    const int row = m0 + wm + j + 8 * half;
    const size_t base = ((size_t)b * S_ + row) * HID_ + h * DHEAD_;
    ctx[base + colA] = f2h_u16(c0[j]);
    ctx[base + colB] = f2h_u16(c1[j]);
  }
}

// ---------------------------------------------------------------------------
extern "C" void kernel_launch(void* const* d_in, const int* in_sizes, int n_in,
                              void* d_out, int out_size, void* d_ws,
                              size_t ws_size, hipStream_t stream) {
  (void)in_sizes; (void)n_in; (void)out_size; (void)ws_size;

  const float* query = (const float*)d_in[0];
  const float* key_i = (const float*)d_in[1];
  const float* value = (const float*)d_in[2];
  const float* Wq = (const float*)d_in[3];  const float* bq = (const float*)d_in[4];
  const float* Wk = (const float*)d_in[5];  const float* bk = (const float*)d_in[6];
  const float* Wv = (const float*)d_in[7];  const float* bv = (const float*)d_in[8];
  const float* Wo = (const float*)d_in[9];  const float* bo = (const float*)d_in[10];
  const int*  mask = (const int*)d_in[11];

  float* x    = (float*)d_out;                            // (B,S,HID)
  float* attn = x + (size_t)B_ * S_ * HID_;               // (B,H,S,S)

  const size_t NTOK = (size_t)B_ * S_ * HID_;             // 8M elements
  unsigned short* Qws = (unsigned short*)d_ws;            // f16 workspace
  unsigned short* Kws = Qws + NTOK;
  unsigned short* Vws = Kws + NTOK;
  unsigned short* Cws = Vws + NTOK;                       // 64 MB total

  const dim3 blk(256);
  const dim3 gProj(HID_ / 64, (B_ * S_) / 64);            // (16, 128)

  gemm_bias_wmma<false, true><<<gProj, blk, 0, stream>>>(query, Wq, bq, Qws,
                                                         B_ * S_, HID_, HID_);
  gemm_bias_wmma<false, true><<<gProj, blk, 0, stream>>>(key_i, Wk, bk, Kws,
                                                         B_ * S_, HID_, HID_);
  gemm_bias_wmma<false, true><<<gProj, blk, 0, stream>>>(value, Wv, bv, Vws,
                                                         B_ * S_, HID_, HID_);

  attn_scores_softmax_wmma<<<dim3(S_ / 16, B_ * HEADS_), blk, 0, stream>>>(
      (const unsigned int*)Qws, (const unsigned int*)Kws, mask, attn);

  attn_v_wmma<<<dim3(S_ / 64, B_ * HEADS_), blk, 0, stream>>>(
      attn, (const unsigned int*)Vws, Cws);

  gemm_bias_wmma<true, false><<<gProj, blk, 0, stream>>>(Cws, Wo, bo, x,
                                                         B_ * S_, HID_, HID_);
}
